// SelfAttention_39676907881763
// MI455X (gfx1250) — matrix operand order repaired and checked by reading, not compiled, not run on hardware
//
#include <hip/hip_runtime.h>

typedef __bf16 bf16_t;
typedef __attribute__((ext_vector_type(16))) __bf16 v16bf;
typedef __attribute__((ext_vector_type(8)))  float  v8f;

union FragBF {
    v16bf v;
    uint4 u[2];
};

// ---------------------------------------------------------------------------
// fp32 -> bf16 elementwise conversion
// ---------------------------------------------------------------------------
__global__ void k_f32_to_bf16(const float* __restrict__ in,
                              bf16_t* __restrict__ out, long n) {
    long i = (long)blockIdx.x * blockDim.x + threadIdx.x;
    long stride = (long)gridDim.x * blockDim.x;
    for (; i < n; i += stride) out[i] = (bf16_t)in[i];
}

// ---------------------------------------------------------------------------
// Row softmax over last dim (ncols == 1024), in-place on bf16.
// ---------------------------------------------------------------------------
__global__ void k_softmax_rows(bf16_t* __restrict__ Q, int ncols) {
    const int t = threadIdx.x;          // 256 threads
    bf16_t* rp = Q + (long)blockIdx.x * ncols;
    float v[4];
    float m = -3.4e38f;
#pragma unroll
    for (int i = 0; i < 4; ++i) {
        v[i] = (float)rp[t + i * 256];
        m = fmaxf(m, v[i]);
    }
    __shared__ float red[256];
    red[t] = m;
    __syncthreads();
    for (int s = 128; s > 0; s >>= 1) {
        if (t < s) red[t] = fmaxf(red[t], red[t + s]);
        __syncthreads();
    }
    m = red[0];
    __syncthreads();
    float sum = 0.0f;
#pragma unroll
    for (int i = 0; i < 4; ++i) {
        v[i] = __expf(v[i] - m);
        sum += v[i];
    }
    red[t] = sum;
    __syncthreads();
    for (int s = 128; s > 0; s >>= 1) {
        if (t < s) red[t] += red[t + s];
        __syncthreads();
    }
    const float inv = 1.0f / red[0];
#pragma unroll
    for (int i = 0; i < 4; ++i) rp[t + i * 256] = (bf16_t)(v[i] * inv);
}

// ---------------------------------------------------------------------------
// Column softmax over sequence dim (axis=-2), in-place on bf16 [B, n, d].
// Block: 256 threads = 64 columns x 4 sequence chunks. grid=(d/64, B).
// ---------------------------------------------------------------------------
__global__ void k_softmax_cols(bf16_t* __restrict__ K, int n, int d) {
    const int t = threadIdx.x;
    const int col = blockIdx.x * 64 + (t & 63);
    const int chunk = t >> 6;                 // 0..3
    const int seg = n >> 2;
    bf16_t* Kp = K + (long)blockIdx.y * n * d;
    const int n0 = chunk * seg, n1 = n0 + seg;

    float m = -3.4e38f;
    for (int i = n0; i < n1; ++i)
        m = fmaxf(m, (float)Kp[(long)i * d + col]);

    __shared__ float red[4][64];
    __shared__ float colM[64];
    __shared__ float colS[64];
    red[chunk][t & 63] = m;
    __syncthreads();
    if (chunk == 0)
        colM[t & 63] = fmaxf(fmaxf(red[0][t & 63], red[1][t & 63]),
                             fmaxf(red[2][t & 63], red[3][t & 63]));
    __syncthreads();
    const float M = colM[t & 63];

    float s = 0.0f;
    for (int i = n0; i < n1; ++i)
        s += __expf((float)Kp[(long)i * d + col] - M);
    red[chunk][t & 63] = s;
    __syncthreads();
    if (chunk == 0)
        colS[t & 63] = red[0][t & 63] + red[1][t & 63] +
                       red[2][t & 63] + red[3][t & 63];
    __syncthreads();
    const float inv = 1.0f / colS[t & 63];

    for (int i = n0; i < n1; ++i) {
        long idx = (long)i * d + col;
        Kp[idx] = (bf16_t)(__expf((float)Kp[idx] - M) * inv);
    }
}

// ---------------------------------------------------------------------------
// bf16 WMMA GEMM:  C = alpha * op(A) * B  (+ bias)
//   op(A) = A    (TRANS_A == false), A is [M x K], row stride lda
//   op(A) = A^T  (TRANS_A == true),  A is [K x M] in memory, row stride lda
//   B is [K x N], row stride ldb. C row stride = N.
// Block: 256 threads (8 waves), 128x128 tile, 32-deep K steps.
// Double-buffered LDS; A tile (non-transposed) staged via
// GLOBAL_LOAD_ASYNC_TO_LDS_B128 with split s_wait_asynccnt, so phase p+1
// async copies overlap phase p WMMA compute. B (and transposed A) tiles
// are register-prefetched one phase ahead.
// ---------------------------------------------------------------------------
template <bool TRANS_A, bool WITH_BIAS, bool OUT_F32>
__global__ __launch_bounds__(256)
void k_gemm_bf16(const bf16_t* __restrict__ A, const bf16_t* __restrict__ B,
                 const float* __restrict__ bias,
                 bf16_t* __restrict__ Cb, float* __restrict__ Cf,
                 int M, int N, int K, int lda, int ldb, float alpha,
                 long sA, long sB, long sC) {
    __shared__ __align__(16) bf16_t lA[2][128][40];   // [m][k] rows of 32 + pad
    __shared__ __align__(16) bf16_t lB[2][128][40];   // [n][k] (B transposed)

    const int t = threadIdx.x;
    const int lane = t & 31;
    const int wave = t >> 5;
    const int waveM = (wave >> 1) * 32;   // 0,32,64,96
    const int waveN = (wave & 1) * 64;    // 0,64
    const int m0 = blockIdx.y * 128;
    const int n0 = blockIdx.x * 128;

    const bf16_t* Ag = A + (long)blockIdx.z * sA;
    const bf16_t* Bg = B + (long)blockIdx.z * sB;

    v8f acc[2][4];
    const v8f vzero = {};
#pragma unroll
    for (int i = 0; i < 2; ++i)
#pragma unroll
        for (int j = 0; j < 4; ++j) acc[i][j] = vzero;

    // ---- staging helpers (fixed per-thread chunk coordinates) ----
    auto issueA = [&](int k0, int buf) {     // async global -> LDS, 2 x 16B
#pragma unroll
        for (int h = 0; h < 2; ++h) {
            int c = t + h * 256;             // 0..511
            int r = c >> 2;                  // 0..127
            int jc = (c & 3) << 3;           // 0,8,16,24
            unsigned lds = (unsigned)(uintptr_t)(&lA[buf][r][jc]);
            unsigned long long ga =
                (unsigned long long)(uintptr_t)(Ag + (long)(m0 + r) * lda + k0 + jc);
            asm volatile("global_load_async_to_lds_b128 %0, %1, off"
                         :: "v"(lds), "v"(ga) : "memory");
        }
    };
    auto loadAt = [&](int k0, uint4* au) {   // TRANS_A: batch global loads
#pragma unroll
        for (int h = 0; h < 2; ++h) {
            int c = t + h * 256;
            int j = c >> 4;                  // k within tile, 0..31
            int ic = (c & 15) << 3;          // m within tile, 0..120
            au[h] = *reinterpret_cast<const uint4*>(
                Ag + (long)(k0 + j) * lda + m0 + ic);
        }
    };
    auto scatterAt = [&](const uint4* au, int buf) {
#pragma unroll
        for (int h = 0; h < 2; ++h) {
            int c = t + h * 256;
            int j = c >> 4, ic = (c & 15) << 3;
            const bf16_t* e = reinterpret_cast<const bf16_t*>(&au[h]);
#pragma unroll
            for (int q = 0; q < 8; ++q) lA[buf][ic + q][j] = e[q];
        }
    };
    auto loadB = [&](int k0, uint4* bu) {
#pragma unroll
        for (int h = 0; h < 2; ++h) {
            int c = t + h * 256;
            int kk = c >> 4;                 // 0..31
            int nc = (c & 15) << 3;          // 0..120
            bu[h] = *reinterpret_cast<const uint4*>(
                Bg + (long)(k0 + kk) * ldb + n0 + nc);
        }
    };
    auto scatterB = [&](const uint4* bu, int buf) {
#pragma unroll
        for (int h = 0; h < 2; ++h) {
            int c = t + h * 256;
            int kk = c >> 4, nc = (c & 15) << 3;
            const bf16_t* e = reinterpret_cast<const bf16_t*>(&bu[h]);
#pragma unroll
            for (int q = 0; q < 8; ++q) lB[buf][nc + q][kk] = e[q];
        }
    };

    const int nsteps = K >> 5;               // K / 32
    uint4 bcur[2], bnxt[2];
    uint4 acur[2], anxt[2];

    // ---- prologue: phase 0 in flight ----
    if (!TRANS_A) issueA(0, 0);
    else          loadAt(0, acur);
    loadB(0, bcur);

    for (int p = 0; p < nsteps; ++p) {
        const int buf = p & 1;
        const int k0 = p << 5;
        const bool more = (p + 1) < nsteps;

        // commit prefetched tiles of phase p to LDS
        scatterB(bcur, buf);
        if (TRANS_A) scatterAt(acur, buf);

        // launch phase p+1 fetches (overlap with this phase's compute)
        if (more) {
            if (!TRANS_A) issueA(k0 + 32, buf ^ 1);
            else          loadAt(k0 + 32, anxt);
            loadB(k0 + 32, bnxt);
        }

        if (!TRANS_A) {
            // asyncs complete in order: <=2 leaves only phase p+1's copies
            if (more) asm volatile("s_wait_asynccnt 0x2" ::: "memory");
            else      asm volatile("s_wait_asynccnt 0x0" ::: "memory");
        }
        __syncthreads();

        // ---- fragments per ISA 16-bit A/B VGPR layouts ----
        FragBF fa[2], fb[4];
        const int sub = lane & 15;
        const int akb = (lane >> 4) << 3;    // A: K base 0 or 8
#pragma unroll
        for (int tm = 0; tm < 2; ++tm) {
            int r = waveM + tm * 16 + sub;
            fa[tm].u[0] = *reinterpret_cast<const uint4*>(&lA[buf][r][akb]);
            fa[tm].u[1] = *reinterpret_cast<const uint4*>(&lA[buf][r][akb + 16]);
        }
        const int bkb = (lane >> 4) << 3;    // B: K base 0 or 8
#pragma unroll
        for (int tn = 0; tn < 4; ++tn) {
            int cN = waveN + tn * 16 + sub;
            fb[tn].u[0] = *reinterpret_cast<const uint4*>(&lB[buf][cN][bkb]);
            fb[tn].u[1] = *reinterpret_cast<const uint4*>(&lB[buf][cN][bkb + 16]);
        }
#pragma unroll
        for (int tm = 0; tm < 2; ++tm)
#pragma unroll
            for (int tn = 0; tn < 4; ++tn)
                acc[tm][tn] = __builtin_amdgcn_wmma_f32_16x16x32_bf16(
                    false, fa[tm].v, false, fb[tn].v,
                    (short)0, acc[tm][tn], false, false);
        __syncthreads();

        if (more) {
#pragma unroll
            for (int h = 0; h < 2; ++h) {
                bcur[h] = bnxt[h];
                if (TRANS_A) acur[h] = anxt[h];
            }
        }
    }

    // ---- epilogue: C/D layout = VGPR r: lanes 0-15 M=r, lanes 16-31 M=r+8 ----
    const int cn = lane & 15;
    const int rmo = (lane >> 4) << 3;
#pragma unroll
    for (int tm = 0; tm < 2; ++tm)
#pragma unroll
        for (int tn = 0; tn < 4; ++tn)
#pragma unroll
            for (int r = 0; r < 8; ++r) {
                int row = m0 + waveM + tm * 16 + rmo + r;
                int col = n0 + waveN + tn * 16 + cn;
                float val = acc[tm][tn][r] * alpha;
                long idx = (long)blockIdx.z * sC + (long)row * N + col;
                if (OUT_F32) {
                    Cf[idx] = val + (WITH_BIAS ? bias[col] : 0.0f);
                } else {
                    Cb[idx] = (bf16_t)val;
                }
            }
}

// ---------------------------------------------------------------------------
// Host-side orchestration
// ---------------------------------------------------------------------------
extern "C" void kernel_launch(void* const* d_in, const int* in_sizes, int n_in,
                              void* d_out, int out_size, void* d_ws, size_t ws_size,
                              hipStream_t stream) {
    const float* x  = (const float*)d_in[0];
    const float* Wq = (const float*)d_in[1];
    const float* Wk = (const float*)d_in[2];
    const float* Wv = (const float*)d_in[3];
    const float* Wo = (const float*)d_in[4];
    const float* bo = (const float*)d_in[5];
    float* out = (float*)d_out;

    const long Bb = 4, Nn = 4096, D = 1024;
    const long MT = Bb * Nn;                 // 16384 total rows

    char* ws = (char*)d_ws;
    size_t off = 0;
    auto alloc = [&](size_t bytes) -> char* {
        char* p = ws + off;
        off += (bytes + 255) & ~(size_t)255;
        return p;
    };
    bf16_t* xb   = (bf16_t*)alloc(MT * D * 2);
    bf16_t* Wqb  = (bf16_t*)alloc(D * D * 2);
    bf16_t* Wkb  = (bf16_t*)alloc(D * D * 2);
    bf16_t* Wvb  = (bf16_t*)alloc(D * D * 2);
    bf16_t* Wob  = (bf16_t*)alloc(D * D * 2);
    bf16_t* Qb   = (bf16_t*)alloc(MT * D * 2);
    bf16_t* Kb   = (bf16_t*)alloc(MT * D * 2);
    bf16_t* Vb   = (bf16_t*)alloc(MT * D * 2);
    bf16_t* ctxb = (bf16_t*)alloc(Bb * D * D * 2);
    bf16_t* yb   = (bf16_t*)alloc(MT * D * 2);
    if (off > ws_size) return;               // insufficient scratch

    // fp32 -> bf16 conversions
    k_f32_to_bf16<<<512, 256, 0, stream>>>(x,  xb,  MT * D);
    k_f32_to_bf16<<<256, 256, 0, stream>>>(Wq, Wqb, D * D);
    k_f32_to_bf16<<<256, 256, 0, stream>>>(Wk, Wkb, D * D);
    k_f32_to_bf16<<<256, 256, 0, stream>>>(Wv, Wvb, D * D);
    k_f32_to_bf16<<<256, 256, 0, stream>>>(Wo, Wob, D * D);

    const dim3 blk(256);
    // Q/K/V projections: [16384 x 1024] * [1024 x 1024]
    k_gemm_bf16<false, false, false><<<dim3(8, 128, 1), blk, 0, stream>>>(
        xb, Wqb, nullptr, Qb, nullptr, 16384, 1024, 1024, 1024, 1024, 1.0f, 0, 0, 0);
    k_gemm_bf16<false, false, false><<<dim3(8, 128, 1), blk, 0, stream>>>(
        xb, Wkb, nullptr, Kb, nullptr, 16384, 1024, 1024, 1024, 1024, 1.0f, 0, 0, 0);
    k_gemm_bf16<false, false, false><<<dim3(8, 128, 1), blk, 0, stream>>>(
        xb, Wvb, nullptr, Vb, nullptr, 16384, 1024, 1024, 1024, 1024, 1.0f, 0, 0, 0);

    // softmax(Q) over features; softmax(K) over sequence
    k_softmax_rows<<<16384, 256, 0, stream>>>(Qb, 1024);
    k_softmax_cols<<<dim3(16, 4), 256, 0, stream>>>(Kb, 4096, 1024);

    // ctx = K^T V * (1/sqrt(D_HEAD)), per batch: [1024x4096]*[4096x1024]
    k_gemm_bf16<true, false, false><<<dim3(8, 8, 4), blk, 0, stream>>>(
        Kb, Vb, nullptr, ctxb, nullptr, 1024, 1024, 4096, 1024, 1024,
        0.125f, (long)4096 * 1024, (long)4096 * 1024, (long)1024 * 1024);

    // y = Q ctx / D_OUT, per batch: [4096x1024]*[1024x1024]
    k_gemm_bf16<false, false, false><<<dim3(8, 32, 4), blk, 0, stream>>>(
        Qb, ctxb, nullptr, yb, nullptr, 4096, 1024, 1024, 1024, 1024,
        1.0f / 1024.0f, (long)4096 * 1024, (long)1024 * 1024, (long)4096 * 1024);

    // out = y Wo + bo (fp32 output)
    k_gemm_bf16<false, true, true><<<dim3(8, 128, 1), blk, 0, stream>>>(
        yb, Wob, bo, nullptr, out, 16384, 1024, 1024, 1024, 1024, 1.0f, 0, 0, 0);
}
